// AssignmentLSTMLanguageModel_30219389895139
// MI455X (gfx1250) — matrix operand order repaired and checked by reading, not compiled, not run on hardware
//
#include <hip/hip_runtime.h>
#include <hip/hip_bf16.h>

// ---------------------------------------------------------------------------
// Types for CDNA5 WMMA (wave32): v_wmma_f32_16x16x32_bf16
//   A: 16x32 bf16  (16 bf16 per lane), B: 32x16 bf16, C/D: 16x16 f32 (8 per lane)
// ---------------------------------------------------------------------------
typedef __attribute__((ext_vector_type(16))) __bf16 v16bf;
typedef __attribute__((ext_vector_type(8)))  float  v8f;

union FragBF {
    v16bf v;
    uint4 u[2];  // two 16B chunks per lane
};

__device__ __forceinline__ unsigned short f32_to_bf16_rne(float f) {
    unsigned int u = __float_as_uint(f);
    u += 0x7fffu + ((u >> 16) & 1u);   // round-to-nearest-even
    return (unsigned short)(u >> 16);
}

// ---------------------------------------------------------------------------
// Constants for this model
// ---------------------------------------------------------------------------
#define BB    16
#define TT    256
#define EE    256
#define HH    512
#define CC    2
#define VV    32000
#define C4H   4096      // C * 4H
#define BT    4096      // B * T

// ---------------------------------------------------------------------------
// Embedding gather (f32 table -> bf16 rows, time-major) + cell-id gather
// ---------------------------------------------------------------------------
__global__ void gather_embed_kernel(const int* __restrict__ x,
                                    const int* __restrict__ assign,
                                    const float* __restrict__ table,
                                    unsigned short* __restrict__ emb_bf,
                                    int* __restrict__ cids) {
    int tb = blockIdx.x;            // t*B + b
    int t  = tb >> 4;
    int b  = tb & 15;
    int tok = x[b * TT + t];        // x is [B,T] row-major
    int e = threadIdx.x;
    emb_bf[(size_t)tb * EE + e] = f32_to_bf16_rne(table[(size_t)tok * EE + e]);
    if (e == 0) cids[tb] = assign[tok];
}

// ---------------------------------------------------------------------------
// Generic f32 -> bf16 conversion (weights)
// ---------------------------------------------------------------------------
__global__ void f32_to_bf16_kernel(const float* __restrict__ in,
                                   unsigned short* __restrict__ out, int n) {
    int i = blockIdx.x * blockDim.x + threadIdx.x;
    if (i < n) out[i] = f32_to_bf16_rne(in[i]);
}

// ---------------------------------------------------------------------------
// Init h/c state (f32 copies + bf16 mirror of h)
// ---------------------------------------------------------------------------
__global__ void init_state_kernel(const float* __restrict__ h0,
                                  const float* __restrict__ c0,
                                  float* __restrict__ h, float* __restrict__ c,
                                  unsigned short* __restrict__ hbf) {
    int i = blockIdx.x * blockDim.x + threadIdx.x;  // 0..8191
    h[i] = h0[i];
    c[i] = c0[i];
    hbf[i] = f32_to_bf16_rne(h0[i]);
}

// ---------------------------------------------------------------------------
// bf16 WMMA GEMM:  D[M,N] = A[M,K](bf16) * W[N,K](bf16)^T + bias[N]
// One wave per block; each wave computes a 32x64 tile:
//   2 M-subtiles x 4 N-subtiles, with each B fragment shared by both M-tiles
//   (halves L2 B-traffic vs 16x64 and gives 2 WMMAs per load-wait).
// K multiple of 32; M multiple of 32; N multiple of 64.
//
// Fragment layouts (wave32, per ISA 7.12.2):
//   A lane: m = lane&15, half = lane>>4; elem[i] -> K = k0 + 16*(i>>3) + 8*half + (i&7)
//           => two contiguous 16B loads per lane.
//   B lane: n = lane&15, half = lane>>4; elem[i] -> K = k0 + 16*(i>>3) + 8*half + (i&7)
//           => two contiguous 16B loads per lane from row-major [N,K] weights.
//   D lane: (vgpr v, lane l) -> M = v + 8*(l>>4), N = l&15
// ---------------------------------------------------------------------------
__global__ void gemm_bf16_kernel(const unsigned short* __restrict__ A,
                                 const unsigned short* __restrict__ W,
                                 const float* __restrict__ bias,
                                 float* __restrict__ D,
                                 int K, int N) {
    const int lane = threadIdx.x;          // 0..31
    const int half = lane >> 4;
    const int lm   = lane & 15;
    const int m0   = blockIdx.y * 32;
    const int n0   = blockIdx.x * 64;

    v8f acc[2][4] = {};
    const unsigned short* Arow0 = A + (size_t)(m0 + lm) * K;
    const unsigned short* Arow1 = Arow0 + (size_t)16 * K;

    for (int k0 = 0; k0 < K; k0 += 32) {
        FragBF a0, a1;
        a0.u[0] = *(const uint4*)(Arow0 + k0 + 8 * half);
        a0.u[1] = *(const uint4*)(Arow0 + k0 + 16 + 8 * half);
        a1.u[0] = *(const uint4*)(Arow1 + k0 + 8 * half);
        a1.u[1] = *(const uint4*)(Arow1 + k0 + 16 + 8 * half);
#pragma unroll
        for (int j = 0; j < 4; ++j) {
            FragBF bf;
            const unsigned short* Wr =
                W + (size_t)(n0 + 16 * j + lm) * K + k0 + 8 * half;
            bf.u[0] = ((const uint4*)Wr)[0];
            bf.u[1] = *(const uint4*)(Wr + 16);
            acc[0][j] = __builtin_amdgcn_wmma_f32_16x16x32_bf16(
                false, a0.v, false, bf.v, (short)0, acc[0][j], false, false);
            acc[1][j] = __builtin_amdgcn_wmma_f32_16x16x32_bf16(
                false, a1.v, false, bf.v, (short)0, acc[1][j], false, false);
        }
    }

#pragma unroll
    for (int j = 0; j < 4; ++j) {
        int n = n0 + 16 * j + lm;
        float bv = bias ? bias[n] : 0.f;
#pragma unroll
        for (int v = 0; v < 8; ++v) {
            D[(size_t)(m0 + v + 8 * half) * N + n]      = acc[0][j][v] + bv;
            D[(size_t)(m0 + 16 + v + 8 * half) * N + n] = acc[1][j][v] + bv;
        }
    }
}

// ---------------------------------------------------------------------------
// One LSTM timestep:
//   gates(c,b,:) = h @ W_hh[c]^T + Xg[t,b,c,:]   (Xg already has x@W_ih^T + b)
//   select by cid(b), apply i/f/g/o nonlinearities, update h,c
// grid: 32 blocks (16 hidden cols each), 256 threads = 8 waves.
// wave w -> (cell = w>>2, gate q = w&3) computes one 16x16 tile (M=batch).
// h_bf is double-buffered across timesteps (all blocks read full h).
// ---------------------------------------------------------------------------
__global__ void lstm_step_kernel(int t,
                                 const float* __restrict__ Xg,
                                 const unsigned short* __restrict__ Whh,
                                 const int* __restrict__ cids,
                                 float* __restrict__ h_f32,
                                 float* __restrict__ c_f32,
                                 const unsigned short* __restrict__ hbf_in,
                                 unsigned short* __restrict__ hbf_out,
                                 unsigned short* __restrict__ hs_bf) {
    __shared__ float gsh[8][BB * 16];   // [wave][batch*16 + hh]

    const int tid  = threadIdx.x;
    const int w    = tid >> 5;
    const int lane = tid & 31;
    const int half = lane >> 4;
    const int lm   = lane & 15;
    const int cell = w >> 2;
    const int q    = w & 3;
    const int hbase = blockIdx.x * 16;
    const int n0    = cell * 2048 + q * HH + hbase;   // column into [C*4H]

    v8f acc = {};
    const unsigned short* Arow = hbf_in + (size_t)lm * HH;   // A row m = batch
    const unsigned short* Wr   = Whh + (size_t)(n0 + lm) * HH + 8 * half;
#pragma unroll
    for (int k0 = 0; k0 < HH; k0 += 32) {
        FragBF a, bf;
        a.u[0]  = *(const uint4*)(Arow + k0 + 8 * half);
        a.u[1]  = *(const uint4*)(Arow + k0 + 16 + 8 * half);
        bf.u[0] = ((const uint4*)(Wr + k0))[0];
        bf.u[1] = ((const uint4*)(Wr + k0 + 16))[0];
        acc = __builtin_amdgcn_wmma_f32_16x16x32_bf16(
            false, a.v, false, bf.v, (short)0, acc, false, false);
    }

#pragma unroll
    for (int v = 0; v < 8; ++v) {
        int b = v + 8 * half;
        gsh[w][b * 16 + lm] =
            acc[v] + Xg[(size_t)(t * BB + b) * C4H + (n0 + lm)];
    }
    __syncthreads();

    // pointwise: one thread per (batch, hh)
    int b  = tid >> 4;
    int hh = tid & 15;
    int cid = cids[t * BB + b];
    int gb  = cid * 4;
    float gi = gsh[gb + 0][b * 16 + hh];
    float gf = gsh[gb + 1][b * 16 + hh];
    float gg = gsh[gb + 2][b * 16 + hh];
    float go = gsh[gb + 3][b * 16 + hh];

    int   hidx = hbase + hh;
    float cOld = c_f32[b * HH + hidx];
    float si = 1.f / (1.f + __expf(-gi));
    float sf = 1.f / (1.f + __expf(-gf));
    float so = 1.f / (1.f + __expf(-go));
    float c2 = sf * cOld + si * tanhf(gg);
    float h2 = so * tanhf(c2);

    c_f32[b * HH + hidx] = c2;
    h_f32[b * HH + hidx] = h2;
    unsigned short hb = f32_to_bf16_rne(h2);
    hbf_out[b * HH + hidx] = hb;
    hs_bf[((size_t)b * TT + t) * HH + hidx] = hb;   // row r = b*T + t
}

// ---------------------------------------------------------------------------
// In-place row-wise log-softmax over N=32000, one block per row
// ---------------------------------------------------------------------------
__global__ void log_softmax_kernel(float* __restrict__ logits, int N) {
    __shared__ float red[256];
    float* p = logits + (size_t)blockIdx.x * N;
    int tid = threadIdx.x;

    float m = -3.402823466e38f;
    for (int i = tid; i < N; i += 256) m = fmaxf(m, p[i]);
    red[tid] = m; __syncthreads();
    for (int s = 128; s > 0; s >>= 1) {
        if (tid < s) red[tid] = fmaxf(red[tid], red[tid + s]);
        __syncthreads();
    }
    m = red[0]; __syncthreads();

    float sum = 0.f;
    for (int i = tid; i < N; i += 256) sum += __expf(p[i] - m);
    red[tid] = sum; __syncthreads();
    for (int s = 128; s > 0; s >>= 1) {
        if (tid < s) red[tid] += red[tid + s];
        __syncthreads();
    }
    float lse = m + __logf(red[0]);
    for (int i = tid; i < N; i += 256) p[i] -= lse;
}

// ---------------------------------------------------------------------------
// Append hT, cT after the logits in d_out
// ---------------------------------------------------------------------------
__global__ void copy_hc_kernel(const float* __restrict__ h,
                               const float* __restrict__ c,
                               float* __restrict__ out) {
    int i = blockIdx.x * blockDim.x + threadIdx.x;   // 0..8191
    out[i]              = h[i];
    out[BB * HH + i]    = c[i];
}

// ---------------------------------------------------------------------------
extern "C" void kernel_launch(void* const* d_in, const int* in_sizes, int n_in,
                              void* d_out, int out_size, void* d_ws, size_t ws_size,
                              hipStream_t stream) {
    (void)in_sizes; (void)n_in; (void)out_size; (void)ws_size;

    const int*   x      = (const int*)  d_in[0];
    const int*   assign = (const int*)  d_in[1];
    const float* h0     = (const float*)d_in[2];
    const float* c0     = (const float*)d_in[3];
    const float* table  = (const float*)d_in[4];
    const float* Wih    = (const float*)d_in[5];   // [C,4H,E] -> [C4H, E]
    const float* Whh    = (const float*)d_in[6];   // [C,4H,H] -> [C4H, H]
    const float* bg     = (const float*)d_in[7];   // [C,4H]   -> [C4H]
    const float* Wout   = (const float*)d_in[8];   // [V,H]
    const float* bout   = (const float*)d_in[9];   // [V]
    float* out = (float*)d_out;

    // ---- workspace carve-out (~112 MB) ----
    char* w = (char*)d_ws;
    auto carve = [&](size_t bytes) {
        char* p = w;
        w += (bytes + 255) & ~(size_t)255;
        return p;
    };
    unsigned short* emb_bf  = (unsigned short*)carve((size_t)BT * EE * 2);
    int*            cids    = (int*)           carve((size_t)BT * 4);
    unsigned short* Wih_bf  = (unsigned short*)carve((size_t)C4H * EE * 2);
    unsigned short* Whh_bf  = (unsigned short*)carve((size_t)C4H * HH * 2);
    unsigned short* Wout_bf = (unsigned short*)carve((size_t)VV * HH * 2);
    float*          Xg      = (float*)         carve((size_t)BT * C4H * 4);
    float*          h_f32   = (float*)         carve((size_t)BB * HH * 4);
    float*          c_f32   = (float*)         carve((size_t)BB * HH * 4);
    unsigned short* hbf0    = (unsigned short*)carve((size_t)BB * HH * 2);
    unsigned short* hbf1    = (unsigned short*)carve((size_t)BB * HH * 2);
    unsigned short* hs_bf   = (unsigned short*)carve((size_t)BT * HH * 2);

    // 1. embedding + cell-id gather (f32 -> bf16)
    gather_embed_kernel<<<BT, EE, 0, stream>>>(x, assign, table, emb_bf, cids);

    // 2. weight conversions f32 -> bf16
    f32_to_bf16_kernel<<<(C4H * EE + 255) / 256, 256, 0, stream>>>(Wih, Wih_bf, C4H * EE);
    f32_to_bf16_kernel<<<(C4H * HH + 255) / 256, 256, 0, stream>>>(Whh, Whh_bf, C4H * HH);
    f32_to_bf16_kernel<<<(VV * HH + 255) / 256, 256, 0, stream>>>(Wout, Wout_bf, VV * HH);

    // 3. init state
    init_state_kernel<<<(BB * HH + 255) / 256, 256, 0, stream>>>(h0, c0, h_f32, c_f32, hbf0);

    // 4. hoisted input projection for all timesteps + bias:
    //    Xg[t*B+b, c*2048+g] = emb @ W_ih^T + b      (M=4096, K=256, N=4096)
    gemm_bf16_kernel<<<dim3(C4H / 64, BT / 32), 32, 0, stream>>>(
        emb_bf, Wih_bf, bg, Xg, EE, C4H);

    // 5. sequential LSTM recurrence: 256 small fused kernels (stream-ordered)
    for (int t = 0; t < TT; ++t) {
        const unsigned short* hin  = (t & 1) ? hbf1 : hbf0;
        unsigned short*       hout = (t & 1) ? hbf0 : hbf1;
        lstm_step_kernel<<<HH / 16, 256, 0, stream>>>(
            t, Xg, Whh_bf, cids, h_f32, c_f32, hin, hout, hs_bf);
    }

    // 6. output projection: logits = hs @ W_out^T + b_out  (M=4096, K=512, N=32000)
    gemm_bf16_kernel<<<dim3(VV / 64, BT / 32), 32, 0, stream>>>(
        hs_bf, Wout_bf, bout, out, HH, VV);

    // 7. in-place log-softmax over vocab
    log_softmax_kernel<<<BT, 256, 0, stream>>>(out, VV);

    // 8. append hT, cT
    copy_hc_kernel<<<(BB * HH + 255) / 256, 256, 0, stream>>>(
        h_f32, c_f32, out + (size_t)BT * VV);
}
